// AlignmentModel_36567351558876
// MI455X (gfx1250) — matrix operand order repaired and checked by reading, not compiled, not run on hardware
//
#include <hip/hip_runtime.h>

typedef __attribute__((ext_vector_type(16))) __bf16 bf16x16;
typedef __attribute__((ext_vector_type(8)))  __bf16 bf16x8;
typedef __attribute__((ext_vector_type(4)))  __bf16 bf16x4;
typedef __attribute__((ext_vector_type(8)))  float  f32x8;
typedef __attribute__((ext_vector_type(4)))  float  f32x4;

#define HID   1000   // hidden dim (H)
#define CTX   2000   // context dim (C = GEMM K)
#define BATCH 64
#define SEQ   1024
#define MT    32     // (b,n) rows per workgroup
#define KP    2016   // K padded to multiple of 32 (zero-filled pad)
#define HP    1024   // H padded to multiple of 64 (zero-filled pad)
#define HT    128    // h columns per tile iteration (16 per wave, 8 waves)
#define NHT   8      // HP / HT
#define UBF_OFF    ((size_t)262144)                     // 256 KB, aligned
#define UBF_BYTES  ((size_t)HP * KP * 2)                // padded bf16 U_a

// branch-free tanh: 1 - 2/(exp(2x)+1); saturates correctly at +/-inf
__device__ __forceinline__ float fast_tanh(float x) {
  float e2 = __expf(2.f * x);
  return 1.f - 2.f / (e2 + 1.f);
}

// ---------------------------------------------------------------------------
// Kernel 1: Ws[b,h] = sum_d s_prev[b,d] * W_a[h,d]   (64x1000, trivial FLOPs)
// ---------------------------------------------------------------------------
__global__ __launch_bounds__(256) void ws_gemm(const float* __restrict__ s_prev,
                                               const float* __restrict__ W_a,
                                               float* __restrict__ Ws) {
  int idx = blockIdx.x * 256 + threadIdx.x;
  if (idx >= BATCH * HID) return;
  int b = idx / HID;
  int h = idx - b * HID;
  const float* sp = s_prev + (size_t)b * HID;
  const float* wr = W_a + (size_t)h * HID;
  float acc = 0.f;
#pragma unroll 4
  for (int d = 0; d < HID; d += 4) {
    f32x4 a = *(const f32x4*)(sp + d);
    f32x4 w = *(const f32x4*)(wr + d);
    acc += a[0] * w[0] + a[1] * w[1] + a[2] * w[2] + a[3] * w[3];
  }
  Ws[idx] = acc;
}

// ---------------------------------------------------------------------------
// Kernel 1b: pre-convert U_a (HIDxCTX fp32) -> padded bf16 image (HPxKP),
// zero-filled in both H tail [1000,1024) and K tail [2000,2016).
// ---------------------------------------------------------------------------
__global__ __launch_bounds__(256) void cvt_ua(const float* __restrict__ Ua,
                                              __bf16* __restrict__ Ubf) {
  int idx = blockIdx.x * 256 + threadIdx.x;   // one bf16x4 per thread
  int e = idx << 2;
  if (e >= HP * KP) return;
  int r = e / KP;
  int c = e - r * KP;
  bf16x4 p;
  if (r < HID && c < CTX) {                   // c%4==0 and CTX%4==0 -> full
    f32x4 v = *(const f32x4*)(Ua + (size_t)r * CTX + c);
    p[0] = (__bf16)v[0]; p[1] = (__bf16)v[1];
    p[2] = (__bf16)v[2]; p[3] = (__bf16)v[3];
  } else {
    p[0] = (__bf16)0.f; p[1] = (__bf16)0.f;
    p[2] = (__bf16)0.f; p[3] = (__bf16)0.f;
  }
  *(bf16x4*)(Ubf + (size_t)r * KP + c) = p;
}

// ---------------------------------------------------------------------------
// Shared A-staging: 32 rows x 2000 fp32 h_j -> bf16 LDS, K-pad zero-filled
// ---------------------------------------------------------------------------
__device__ __forceinline__ void stage_A(const float* __restrict__ ab,
                                        __bf16* As, int tid) {
  for (int i = tid; i < MT * (CTX / 4); i += 256) {
    int e = i << 2;
    int r = e / CTX;
    int c = e - r * CTX;
    f32x4 v = *(const f32x4*)(ab + (size_t)r * CTX + c);
    bf16x4 p;
    p[0] = (__bf16)v[0]; p[1] = (__bf16)v[1];
    p[2] = (__bf16)v[2]; p[3] = (__bf16)v[3];
    *(bf16x4*)(&As[r * KP + c]) = p;
  }
  for (int i = tid; i < MT * ((KP - CTX) / 4); i += 256) {
    int r = i >> 2;
    int c = CTX + ((i & 3) << 2);
    bf16x4 z;
    z[0] = (__bf16)0.f; z[1] = (__bf16)0.f; z[2] = (__bf16)0.f; z[3] = (__bf16)0.f;
    *(bf16x4*)(&As[r * KP + c]) = z;
  }
}

__device__ __forceinline__ bf16x16 load_afrag(const __bf16* arow, int ka) {
  bf16x8 lo = *(const bf16x8*)(arow + ka);
  bf16x8 hi = *(const bf16x8*)(arow + ka + 8);
  bf16x16 f;
#pragma unroll
  for (int i = 0; i < 8; ++i) { f[i] = lo[i]; f[8 + i] = hi[i]; }
  return f;
}

// ---------------------------------------------------------------------------
// Kernel 2 (fast path): each wave owns BOTH 16-row fragments (2 accumulators)
// and a private 16-column strip -> every B fragment feeds 2 WMMAs, no
// intra-workgroup B duplication, no conversions in the hot loop.
// ---------------------------------------------------------------------------
__global__ __launch_bounds__(256) void attn_scores_pre(
    const float*  __restrict__ hj,   // (B*SEQ, CTX) fp32
    const __bf16* __restrict__ Ubf,  // (HP, KP) bf16, zero-padded
    const float*  __restrict__ va,   // (HID) fp32
    const float*  __restrict__ Ws,   // (B, HID) fp32
    float* __restrict__ out)         // (B*SEQ) fp32
{
  __shared__ __bf16 As[MT * KP];     // 126 KB
  __shared__ float scores[MT];

  const int tid    = threadIdx.x;
  const int m_base = blockIdx.x * MT;
  const int b      = m_base >> 10;

  if (tid < MT) scores[tid] = 0.f;
  stage_A(hj + (size_t)m_base * CTX, As, tid);
  __syncthreads();

  const int lane = tid & 31;
  const int wave = tid >> 5;         // 8 waves: private 16-col strip each
  const int nl   = lane & 15;
  const int half = lane >> 4;

  // A frag (this code's k order, matched by B; not the ISA table order): lane row = nl; K = half*16 .. +15 contiguous
  const __bf16* arow0 = &As[(nl)      * KP];   // rows 0..15
  const __bf16* arow1 = &As[(16 + nl) * KP];   // rows 16..31

  float sacc0[8], sacc1[8];
#pragma unroll
  for (int j = 0; j < 8; ++j) { sacc0[j] = 0.f; sacc1[j] = 0.f; }

  for (int ht = 0; ht < NHT; ++ht) {
    const int h = ht * HT + wave * 16 + nl;        // < HP, row valid (padded)
    const __bf16* brow = Ubf + (size_t)h * KP;

    f32x8 acc0 = {0.f, 0.f, 0.f, 0.f, 0.f, 0.f, 0.f, 0.f};
    f32x8 acc1 = {0.f, 0.f, 0.f, 0.f, 0.f, 0.f, 0.f, 0.f};

    for (int k = 0; k < KP; k += 32) {
      const int ka = k + half * 16;
      bf16x16 a0 = load_afrag(arow0, ka);
      bf16x16 a1 = load_afrag(arow1, ka);

      // B frag (same k order as A): lane col = nl; K = k+half*16 .. +15 contiguous
      const int kb = k + half * 16;
      bf16x8 blo = *(const bf16x8*)(brow + kb);
      bf16x8 bhi = *(const bf16x8*)(brow + kb + 8);
      bf16x16 bfrag;
#pragma unroll
      for (int i = 0; i < 8; ++i) { bfrag[i] = blo[i]; bfrag[8 + i] = bhi[i]; }

      acc0 = __builtin_amdgcn_wmma_f32_16x16x32_bf16(false, a0, false, bfrag,
                                                     (short)0, acc0, false, false);
      acc1 = __builtin_amdgcn_wmma_f32_16x16x32_bf16(false, a1, false, bfrag,
                                                     (short)0, acc1, false, false);
    }

    const float wsv = h < HID ? Ws[b * HID + h] : 0.f;
    const float vv  = h < HID ? va[h] : 0.f;       // masks H padding
    float p0[8], p1[8];
#pragma unroll
    for (int j = 0; j < 8; ++j) {
      p0[j] = vv * fast_tanh(acc0[j] + wsv);
      p1[j] = vv * fast_tanh(acc1[j] + wsv);
    }
#pragma unroll
    for (int off = 8; off >= 1; off >>= 1) {
#pragma unroll
      for (int j = 0; j < 8; ++j) {
        p0[j] += __shfl_xor(p0[j], off, 16);
        p1[j] += __shfl_xor(p1[j], off, 16);
      }
    }
#pragma unroll
    for (int j = 0; j < 8; ++j) { sacc0[j] += p0[j]; sacc1[j] += p1[j]; }
  }

  // C/D layout: VGPR j, lane half -> local row j + half*8 (acc1: +16)
  if (nl == 0) {
    const int mr = half * 8;
#pragma unroll
    for (int j = 0; j < 8; ++j) {
      atomicAdd(&scores[mr + j],      sacc0[j]);
      atomicAdd(&scores[16 + mr + j], sacc1[j]);
    }
  }
  __syncthreads();
  if (tid < MT) out[m_base + tid] = scores[tid];
}

// ---------------------------------------------------------------------------
// Kernel 2 (fallback, ws too small): convert fp32 B in the hot loop
// ---------------------------------------------------------------------------
__global__ __launch_bounds__(256) void attn_scores_cvt(
    const float* __restrict__ hj, const float* __restrict__ Ua,
    const float* __restrict__ va, const float* __restrict__ Ws,
    float* __restrict__ out)
{
  __shared__ __bf16 As[MT * KP];
  __shared__ float scores[MT];

  const int tid    = threadIdx.x;
  const int m_base = blockIdx.x * MT;
  const int b      = m_base >> 10;

  if (tid < MT) scores[tid] = 0.f;
  stage_A(hj + (size_t)m_base * CTX, As, tid);
  __syncthreads();

  const int lane = tid & 31;
  const int wave = tid >> 5;
  const int nl   = lane & 15;
  const int half = lane >> 4;
  const __bf16* arow0 = &As[(nl)      * KP];
  const __bf16* arow1 = &As[(16 + nl) * KP];

  float sacc0[8], sacc1[8];
#pragma unroll
  for (int j = 0; j < 8; ++j) { sacc0[j] = 0.f; sacc1[j] = 0.f; }

  for (int ht = 0; ht < NHT; ++ht) {
    const int h  = ht * HT + wave * 16 + nl;
    const int hh = h < HID ? h : HID - 1;
    const float* brow = Ua + (size_t)hh * CTX;

    f32x8 acc0 = {0.f, 0.f, 0.f, 0.f, 0.f, 0.f, 0.f, 0.f};
    f32x8 acc1 = {0.f, 0.f, 0.f, 0.f, 0.f, 0.f, 0.f, 0.f};

    for (int k = 0; k < KP; k += 32) {
      const int ka = k + half * 16;
      bf16x16 a0 = load_afrag(arow0, ka);
      bf16x16 a1 = load_afrag(arow1, ka);

      int cb = k + half * 16;
      cb = cb <= CTX - 16 ? cb : CTX - 16;   // tail clamp; A pad zeroes it
      f32x4 q0 = *(const f32x4*)(brow + cb);
      f32x4 q1 = *(const f32x4*)(brow + cb + 4);
      f32x4 q2 = *(const f32x4*)(brow + cb + 8);
      f32x4 q3 = *(const f32x4*)(brow + cb + 12);
      bf16x16 bfrag;
#pragma unroll
      for (int i = 0; i < 4; ++i) {
        bfrag[i]      = (__bf16)q0[i];
        bfrag[4 + i]  = (__bf16)q1[i];
        bfrag[8 + i]  = (__bf16)q2[i];
        bfrag[12 + i] = (__bf16)q3[i];
      }

      acc0 = __builtin_amdgcn_wmma_f32_16x16x32_bf16(false, a0, false, bfrag,
                                                     (short)0, acc0, false, false);
      acc1 = __builtin_amdgcn_wmma_f32_16x16x32_bf16(false, a1, false, bfrag,
                                                     (short)0, acc1, false, false);
    }

    const float wsv = h < HID ? Ws[b * HID + h] : 0.f;
    const float vv  = h < HID ? va[h] : 0.f;
    float p0[8], p1[8];
#pragma unroll
    for (int j = 0; j < 8; ++j) {
      p0[j] = vv * fast_tanh(acc0[j] + wsv);
      p1[j] = vv * fast_tanh(acc1[j] + wsv);
    }
#pragma unroll
    for (int off = 8; off >= 1; off >>= 1) {
#pragma unroll
      for (int j = 0; j < 8; ++j) {
        p0[j] += __shfl_xor(p0[j], off, 16);
        p1[j] += __shfl_xor(p1[j], off, 16);
      }
    }
#pragma unroll
    for (int j = 0; j < 8; ++j) { sacc0[j] += p0[j]; sacc1[j] += p1[j]; }
  }

  if (nl == 0) {
    const int mr = half * 8;
#pragma unroll
    for (int j = 0; j < 8; ++j) {
      atomicAdd(&scores[mr + j],      sacc0[j]);
      atomicAdd(&scores[16 + mr + j], sacc1[j]);
    }
  }
  __syncthreads();
  if (tid < MT) out[m_base + tid] = scores[tid];
}

// ---------------------------------------------------------------------------
extern "C" void kernel_launch(void* const* d_in, const int* in_sizes, int n_in,
                              void* d_out, int out_size, void* d_ws, size_t ws_size,
                              hipStream_t stream) {
  (void)in_sizes; (void)n_in; (void)out_size;
  const float* s_prev = (const float*)d_in[0];
  const float* h_j    = (const float*)d_in[1];
  const float* W_a    = (const float*)d_in[2];
  const float* U_a    = (const float*)d_in[3];
  const float* v_a    = (const float*)d_in[4];
  float* out = (float*)d_out;

  char*  wsb = (char*)d_ws;
  float* Ws  = (float*)wsb;                       // 256 KB

  ws_gemm<<<(BATCH * HID + 255) / 256, 256, 0, stream>>>(s_prev, W_a, Ws);

  if (ws_size >= UBF_OFF + UBF_BYTES) {
    __bf16* Ubf = (__bf16*)(wsb + UBF_OFF);       // 4 MB padded bf16 U_a
    cvt_ua<<<(HP * KP / 4 + 255) / 256, 256, 0, stream>>>(U_a, Ubf);
    attn_scores_pre<<<(BATCH * SEQ) / MT, 256, 0, stream>>>(h_j, Ubf, v_a, Ws, out);
  } else {
    attn_scores_cvt<<<(BATCH * SEQ) / MT, 256, 0, stream>>>(h_j, U_a, v_a, Ws, out);
  }
}
